// AttentionHead_43568148250997
// MI455X (gfx1250) — compile-verified, matrix operand order checked by reading, not run on hardware
//
#include <hip/hip_runtime.h>

typedef __attribute__((ext_vector_type(16))) __bf16 v16bf;
typedef __attribute__((ext_vector_type(8)))  __bf16 v8bf;
typedef __attribute__((ext_vector_type(8)))  float  v8f;
typedef __attribute__((ext_vector_type(4)))  float  f32x4;

#define HD   128
#define HID  1024
#define BB   4
#define SS   4096
#define BSM  (BB * SS)   // 16384 total rows

__device__ __forceinline__ v8f vzero8() {
  v8f z = {0.f, 0.f, 0.f, 0.f, 0.f, 0.f, 0.f, 0.f};
  return z;
}

__device__ __forceinline__ v16bf cat8(v8bf lo, v8bf hi) {
  return __builtin_shufflevector(lo, hi, 0, 1, 2, 3, 4, 5, 6, 7,
                                         8, 9, 10, 11, 12, 13, 14, 15);
}

// Load a 16-bit 16x32 A/B fragment from row-major memory: per ISA 7.12.2 the
// per-lane data is two contiguous 8-element runs (K = khalf+0..7 and
// K = 16+khalf+0..7)  ->  exactly two *_load_b128 per lane.
__device__ __forceinline__ v16bf frag_ld(const __bf16* row_ptr /* + khalf */) {
  v8bf lo = *(const v8bf*)row_ptr;
  v8bf hi = *(const v8bf*)(row_ptr + 16);
  return cat8(lo, hi);
}

// gfx1250 async DMA: global -> LDS, 128b per lane, tracked by ASYNCcnt.
__device__ __forceinline__ void async_b128(unsigned lds_byte, const __bf16* g) {
  asm volatile("global_load_async_to_lds_b128 %0, %1, off"
               :: "v"(lds_byte), "v"(g) : "memory");
}

// ---------------------------------------------------------------------------
// Kernel 0: weight prep.  wq/wk/wv [HID x HD] f32 -> bf16 transposed [HD x HID]
//           wo [HD x HID] f32 -> bf16 transposed [HID x HD]
// ---------------------------------------------------------------------------
__global__ void prep_weights_kernel(const float* __restrict__ wq,
                                    const float* __restrict__ wk,
                                    const float* __restrict__ wv,
                                    const float* __restrict__ wo,
                                    __bf16* __restrict__ wqt,
                                    __bf16* __restrict__ wkt,
                                    __bf16* __restrict__ wvt,
                                    __bf16* __restrict__ wot) {
  const int idx = blockIdx.x * 256 + threadIdx.x;
  if (idx >= HD * HID) return;
  { // wqt/wkt/wvt[n][k] = w[k][n]
    const int n = idx / HID, k = idx % HID;
    wqt[idx] = (__bf16)wq[(size_t)k * HD + n];
    wkt[idx] = (__bf16)wk[(size_t)k * HD + n];
    wvt[idx] = (__bf16)wv[(size_t)k * HD + n];
  }
  { // wot[n][k] = wo[k][n], n < HID, k < HD
    const int n = idx / HD, k = idx % HD;
    wot[idx] = (__bf16)wo[(size_t)k * HID + n];
  }
}

// ---------------------------------------------------------------------------
// Kernel 1: fused Q/K/V projection.  x[f32] A-frags via 4x b128 + pk-cvt,
// bf16-transposed weights B-frags via 2x b128.  Q scaled by 1/sqrt(HD).
// V is written TRANSPOSED ([batch][d][s]) so attention's P@V B-fragments
// become contiguous b128 loads.
// ---------------------------------------------------------------------------
__global__ void qkv_proj_kernel(const float*  __restrict__ x,
                                const __bf16* __restrict__ wqt,
                                const __bf16* __restrict__ wkt,
                                const __bf16* __restrict__ wvt,
                                __bf16* __restrict__ qo,
                                __bf16* __restrict__ ko,
                                __bf16* __restrict__ vt) {
  const int lane  = threadIdx.x & 31;
  const int wave  = threadIdx.x >> 5;
  const int lm    = lane & 15;
  const int khalf = (lane >> 4) << 3;             // 0 or 8
  const int m0    = (blockIdx.x * 4 + wave) * 16; // row tile
  const int which = blockIdx.y;

  const __bf16* wt = (which == 0) ? wqt : (which == 1) ? wkt : wvt;
  const float scale = (which == 0) ? 0.08838834764831845f : 1.0f; // 1/sqrt(128)

  v8f acc[8];
#pragma unroll
  for (int n = 0; n < 8; ++n) acc[n] = vzero8();

  for (int kb = 0; kb < HID; kb += 32) {
    // A fragment from f32 x: two 8-float contiguous runs -> 4x b128 + pk cvt
    const float* xr = x + (size_t)(m0 + lm) * HID + kb + khalf;
    __builtin_prefetch(xr + 32, 0, 3);
    f32x4 x0 = *(const f32x4*)(xr + 0);
    f32x4 x1 = *(const f32x4*)(xr + 4);
    f32x4 x2 = *(const f32x4*)(xr + 16);
    f32x4 x3 = *(const f32x4*)(xr + 20);
    v16bf a;
#pragma unroll
    for (int j = 0; j < 4; ++j) {
      a[j]      = (__bf16)x0[j];
      a[4 + j]  = (__bf16)x1[j];
      a[8 + j]  = (__bf16)x2[j];
      a[12 + j] = (__bf16)x3[j];
    }

#pragma unroll
    for (int n = 0; n < 8; ++n) {
      const __bf16* wr = wt + (size_t)(n * 16 + lm) * HID + kb + khalf;
      v16bf b = frag_ld(wr);
      acc[n] = __builtin_amdgcn_wmma_f32_16x16x32_bf16(
          false, a, false, b, (short)0, acc[n], false, false);
    }
  }

  if (which == 2) {  // V: store transposed [batch][d][s]
    const int batch = m0 / SS, srow = m0 % SS;
    __bf16* vtb = vt + (size_t)batch * HD * SS;
#pragma unroll
    for (int n = 0; n < 8; ++n)
#pragma unroll
      for (int r = 0; r < 8; ++r)
        vtb[(size_t)(n * 16 + lm) * SS + srow + r + khalf] = (__bf16)acc[n][r];
  } else {
    __bf16* outp = (which == 0) ? qo : ko;
#pragma unroll
    for (int n = 0; n < 8; ++n)
#pragma unroll
      for (int r = 0; r < 8; ++r)
        outp[(size_t)(m0 + r + khalf) * HD + n * 16 + lm] =
            (__bf16)(acc[n][r] * scale);
  }
}

// ---------------------------------------------------------------------------
// Kernel 2: flash attention.  4 waves / workgroup (64 queries) SHARE the K/V
// LDS staging: each 32-key block's 32 async-b128 DMA ops are split 8-per-wave,
// triple-buffered with a 2-block lookahead.  Per iteration:
//   s_wait_asynccnt(own 8)  ->  barrier  ->  issue block i+2  ->  compute i.
// The barrier both publishes block i's copies and closes the reuse hazard on
// buffer (i+2)%3 (its last readers ran in iteration i-1, before this barrier).
// ---------------------------------------------------------------------------
__global__ void __launch_bounds__(128)
flash_attn_kernel(const __bf16* __restrict__ qb,
                  const __bf16* __restrict__ kbuf,
                  const __bf16* __restrict__ vt,
                  const int*    __restrict__ mask,
                  __bf16* __restrict__ hb) {
  __shared__ __attribute__((aligned(16))) __bf16 kst[3 * 32 * HD];  // [key%32][d]
  __shared__ __attribute__((aligned(16))) __bf16 vst[3 * HD * 32];  // [d][key%32]
  __shared__ __attribute__((aligned(16))) __bf16 pbuf[4 * 16 * 32]; // per-wave P

  const int wave  = threadIdx.x >> 5;
  const int lane  = threadIdx.x & 31;
  const int lm    = lane & 15;
  const int khalf = (lane >> 4) << 3;
  const int m0    = (blockIdx.x * 4 + wave) * 16;
  const size_t baseR = (size_t)blockIdx.y * SS * HD;  // row-major Q/K/h base
  const size_t baseT = (size_t)blockIdx.y * HD * SS;  // transposed V base

  const unsigned ksAddr = (unsigned)(size_t)&kst[0];
  const unsigned vsAddr = (unsigned)(size_t)&vst[0];
  __bf16* pb = &pbuf[wave * 16 * 32];

  // per-lane coords for the async tile copies (each op: 32 lanes x 16B = 512B)
  const int kr2 = lane >> 4, kc = lane & 15;  // K tile: 2 rows (256B) per op
  const int vr8 = lane >> 2, vc = lane & 3;   // V tile: 8 rows (64B)  per op

  // issue this wave's quarter (4 K-ops + 4 V-ops) of one 32-key block
  auto issue_block = [&](int kblk, int buf) {
    const __bf16* kg = kbuf + baseR + (size_t)kblk * HD;
    const __bf16* vg = vt + baseT + kblk;
    const unsigned kb0 = ksAddr + (unsigned)buf * (32 * HD * 2);
    const unsigned vb0 = vsAddr + (unsigned)buf * (HD * 32 * 2);
#pragma unroll
    for (int j = 0; j < 4; ++j) {
      const int i = wave * 4 + j;
      async_b128(kb0 + (unsigned)((2 * i + kr2) * HD + kc * 8) * 2,
                 kg + (size_t)(2 * i + kr2) * HD + kc * 8);
      async_b128(vb0 + (unsigned)((8 * i + vr8) * 32 + vc * 8) * 2,
                 vg + (size_t)(8 * i + vr8) * SS + vc * 8);
    }
  };

  // Q fragments resident in VGPRs (2x global_load_b128 each)
  v16bf qf[4];
#pragma unroll
  for (int d = 0; d < 4; ++d)
    qf[d] = frag_ld(qb + baseR + (size_t)(m0 + lm) * HD + d * 32 + khalf);

  v8f accO[8];
#pragma unroll
  for (int n = 0; n < 8; ++n) accO[n] = vzero8();
  float mrow[8], lrow[8];
#pragma unroll
  for (int r = 0; r < 8; ++r) { mrow[r] = -3.0e38f; lrow[r] = 0.0f; }

  // prologue: two blocks in flight
  issue_block(0, 0);
  issue_block(32, 1);

  const int NIT = SS / 32;  // 128
  for (int it = 0; it < NIT; ++it) {
    const int buf = it % 3;
    if (it + 1 < NIT) {  // blocks it, it+1 outstanding (8 ops each, per wave)
      asm volatile("s_wait_asynccnt 0x8" ::: "memory");
    } else {
      asm volatile("s_wait_asynccnt 0x0" ::: "memory");
    }
    __syncthreads();  // block `it` visible from all waves; buffers safe to reuse
    if (it + 2 < NIT) issue_block((it + 2) * 32, (it + 2) % 3);

    const __bf16* kls = &kst[buf * 32 * HD];
    const __bf16* vls = &vst[buf * HD * 32];

    // scores: two 16x16 tiles over this 32-key block (ds_load_b128 fragments)
    v8f s0 = vzero8(), s1 = vzero8();
#pragma unroll
    for (int d = 0; d < 4; ++d) {
      v16bf kf0 = frag_ld(kls + (size_t)lm * HD + d * 32 + khalf);
      v16bf kf1 = frag_ld(kls + (size_t)(16 + lm) * HD + d * 32 + khalf);
      s0 = __builtin_amdgcn_wmma_f32_16x16x32_bf16(false, qf[d], false, kf0,
                                                   (short)0, s0, false, false);
      s1 = __builtin_amdgcn_wmma_f32_16x16x32_bf16(false, qf[d], false, kf1,
                                                   (short)0, s1, false, false);
    }

    // online softmax (rows live in 16-lane groups -> xor-shuffle reductions)
    float alpha[8];
#pragma unroll
    for (int r = 0; r < 8; ++r) {
      float t = fmaxf(s0[r], s1[r]);
#pragma unroll
      for (int off = 1; off < 16; off <<= 1) t = fmaxf(t, __shfl_xor(t, off, 32));
      const float mnew = fmaxf(mrow[r], t);
      alpha[r] = __expf(mrow[r] - mnew);
      const float p0 = __expf(s0[r] - mnew);
      const float p1 = __expf(s1[r] - mnew);
      s0[r] = p0;
      s1[r] = p1;
      float rs = p0 + p1;
#pragma unroll
      for (int off = 1; off < 16; off <<= 1) rs += __shfl_xor(rs, off, 32);
      lrow[r] = alpha[r] * lrow[r] + rs;
      mrow[r] = mnew;
    }
#pragma unroll
    for (int n = 0; n < 8; ++n)
#pragma unroll
      for (int r = 0; r < 8; ++r) accO[n][r] *= alpha[r];

    // restage P through this wave's private LDS slice (same-wave DS ops are
    // in-order; compiler inserts the dscnt wait): C-layout -> A-layout frag
#pragma unroll
    for (int r = 0; r < 8; ++r) {
      pb[(r + khalf) * 32 + lm]      = (__bf16)s0[r];
      pb[(r + khalf) * 32 + 16 + lm] = (__bf16)s1[r];
    }
    v16bf pf = frag_ld(&pb[lm * 32 + khalf]);

    // accO += P (16x32) @ V (32x128); V fragments contiguous in transposed LDS
#pragma unroll
    for (int n = 0; n < 8; ++n) {
      v16bf vf = frag_ld(vls + (size_t)(n * 16 + lm) * 32 + khalf);
      accO[n] = __builtin_amdgcn_wmma_f32_16x16x32_bf16(
          false, pf, false, vf, (short)0, accO[n], false, false);
    }
  }

  // epilogue: normalize, apply head_dim_mask, store bf16
  float inv[8];
#pragma unroll
  for (int r = 0; r < 8; ++r) inv[r] = 1.0f / lrow[r];
#pragma unroll
  for (int n = 0; n < 8; ++n) {
    const float mv = (float)mask[n * 16 + lm];
#pragma unroll
    for (int r = 0; r < 8; ++r)
      hb[baseR + (size_t)(m0 + r + khalf) * HD + n * 16 + lm] =
          (__bf16)(accO[n][r] * inv[r] * mv);
  }
}

// ---------------------------------------------------------------------------
// Kernel 3: y = h(bf16) @ Wo -> f32, using bf16-transposed Wo; all fragment
// loads are b128.  One wave computes a 16x64 output slice.
// ---------------------------------------------------------------------------
__global__ void out_proj_kernel(const __bf16* __restrict__ hb,
                                const __bf16* __restrict__ wot,
                                float* __restrict__ y) {
  const int lane  = threadIdx.x & 31;
  const int lm    = lane & 15;
  const int khalf = (lane >> 4) << 3;
  const int m0    = blockIdx.x * 16;
  const int n0    = blockIdx.y * 64;

  v8f acc[4];
#pragma unroll
  for (int n = 0; n < 4; ++n) acc[n] = vzero8();

#pragma unroll
  for (int d = 0; d < 4; ++d) {
    v16bf a = frag_ld(hb + (size_t)(m0 + lm) * HD + d * 32 + khalf);
#pragma unroll
    for (int n = 0; n < 4; ++n) {
      v16bf b = frag_ld(wot + (size_t)(n0 + n * 16 + lm) * HD + d * 32 + khalf);
      acc[n] = __builtin_amdgcn_wmma_f32_16x16x32_bf16(
          false, a, false, b, (short)0, acc[n], false, false);
    }
  }

#pragma unroll
  for (int n = 0; n < 4; ++n)
#pragma unroll
    for (int r = 0; r < 8; ++r)
      y[(size_t)(m0 + r + khalf) * HID + n0 + n * 16 + lm] = acc[n][r];
}

// ---------------------------------------------------------------------------
extern "C" void kernel_launch(void* const* d_in, const int* in_sizes, int n_in,
                              void* d_out, int out_size, void* d_ws, size_t ws_size,
                              hipStream_t stream) {
  const float* x    = (const float*)d_in[0];
  const int*   mask = (const int*)d_in[1];
  const float* wq   = (const float*)d_in[2];
  const float* wk   = (const float*)d_in[3];
  const float* wv   = (const float*)d_in[4];
  const float* wo   = (const float*)d_in[5];
  float* y = (float*)d_out;

  const size_t act_bytes = (size_t)BSM * HD * 2;  // 4 MB each
  const size_t w_bytes   = (size_t)HD * HID * 2;  // 256 KB each
  char* ws = (char*)d_ws;
  __bf16* qb  = (__bf16*)(ws + 0 * act_bytes);
  __bf16* kb  = (__bf16*)(ws + 1 * act_bytes);
  __bf16* vt  = (__bf16*)(ws + 2 * act_bytes);  // transposed V [b][d][s]
  __bf16* hb  = (__bf16*)(ws + 3 * act_bytes);
  __bf16* wqt = (__bf16*)(ws + 4 * act_bytes);
  __bf16* wkt = (__bf16*)(ws + 4 * act_bytes + 1 * w_bytes);
  __bf16* wvt = (__bf16*)(ws + 4 * act_bytes + 2 * w_bytes);
  __bf16* wot = (__bf16*)(ws + 4 * act_bytes + 3 * w_bytes);

  // 0) transpose + downcast weights (tiny)
  prep_weights_kernel<<<dim3((HD * HID + 255) / 256), 256, 0, stream>>>(
      wq, wk, wv, wo, wqt, wkt, wvt, wot);
  // 1) Q/K/V projections: 1024 M-tiles, 4 waves/block, y-dim selects Q/K/V
  qkv_proj_kernel<<<dim3(BSM / 16 / 4, 3), 128, 0, stream>>>(x, wqt, wkt, wvt,
                                                             qb, kb, vt);
  // 2) flash attention: 4 waves (64 queries) per block sharing K/V staging
  flash_attn_kernel<<<dim3(SS / 64, BB), 128, 0, stream>>>(qb, kb, vt, mask, hb);
  // 3) output projection
  out_proj_kernel<<<dim3(BSM / 16, HID / 64), 32, 0, stream>>>(hb, wot, y);
}
